// Model_33406255628764
// MI455X (gfx1250) — matrix operand order repaired and checked by reading, not compiled, not run on hardware
//
#include <hip/hip_runtime.h>
#include <math.h>

// ---------------- model constants ----------------
static constexpr int kB   = 32;
static constexpr int kL   = 512;
static constexpr int kCIN = 32;
static constexpr int kD   = 512;
static constexpr int kH   = 8;
static constexpr int kDh  = 64;
static constexpr int kFF  = 2048;
static constexpr int kU   = 35;     // FACTOR * ceil(ln(512)) = 5*7
static constexpr int kCLS = 10;
static constexpr int kM   = kB * kL;      // 16384 rows

typedef unsigned int u32;
typedef unsigned short u16;
typedef __attribute__((ext_vector_type(4)))  u32   u32x4;
typedef __attribute__((ext_vector_type(8)))  u32   u32x8;
typedef __attribute__((ext_vector_type(16))) __bf16 v16bf;
typedef __attribute__((ext_vector_type(8)))  float  v8f;
typedef int v4i __attribute__((vector_size(16)));   // matches builtin param type

#define AS1 __attribute__((address_space(1)))
#define AS3 __attribute__((address_space(3)))

#if defined(__gfx1250__) && __has_builtin(__builtin_amdgcn_global_load_async_to_lds_b128)
#define HAVE_ASYNC_LDS 1
#else
#define HAVE_ASYNC_LDS 0
#endif

// ---------------- small device helpers ----------------
__device__ __forceinline__ u16 f2bf(float x) {
  u32 u = __float_as_uint(x);
  u32 r = u + 0x7FFFu + ((u >> 16) & 1u);   // round-to-nearest-even
  return (u16)(r >> 16);
}
__device__ __forceinline__ float gelu_f(float x) {
  return 0.5f * x * (1.0f + erff(x * 0.70710678118654752f));
}
__device__ __forceinline__ u32 hash3(u32 a, u32 b, u32 c) {
  u32 h = a * 747796405u + b * 2654435761u + c * 0x9E3779B9u + 0x85EBCA6Bu;
  h ^= h >> 15; h *= 0x2C1B3C6Du; h ^= h >> 12; h *= 0x297A2D39u; h ^= h >> 15;
  return h;
}
#if HAVE_ASYNC_LDS
__device__ __forceinline__ void async_cp16(const void* g, void* l) {
  __builtin_amdgcn_global_load_async_to_lds_b128(
      (AS1 v4i*)g, (AS3 v4i*)l, 0, 0);
}
__device__ __forceinline__ void async_wait0() {
#if __has_builtin(__builtin_amdgcn_s_wait_asynccnt)
  __builtin_amdgcn_s_wait_asynccnt(0);
#else
  asm volatile("s_wait_asynccnt 0x0" ::: "memory");
#endif
}
#endif

// ---------------- TDM weight preloader (L2 warmer) ----------------
// Each 1-wave workgroup issues one tensor_load_to_lds pulling a 16KB chunk of
// the weight matrix through L2 into its LDS staging buffer, then waits on
// TENSORcnt.  D# per CDNA5 ISA (group0: count/lds/global/
// type=2, group1: data_size=4B, tile 1 x 4096 elements).
__global__ __launch_bounds__(32) void k_tdm_preload(const u32* __restrict__ gsrc) {
  __shared__ __align__(128) u32 staging[4096];   // 16 KB
#if defined(__gfx1250__)
  u32 lds_base = (u32)(unsigned long long)(AS3 u32*)staging;
  unsigned long long ga =
      (unsigned long long)(const void*)(gsrc + (size_t)blockIdx.x * 4096);
  u32x4 g0;
  g0[0] = 1u;                                              // count=1, user mode
  g0[1] = lds_base;                                        // lds_addr
  g0[2] = (u32)ga;                                         // global_addr[31:0]
  g0[3] = (u32)((ga >> 32) & 0x1FFFFFFull) | (2u << 30);   // addr[56:32] | type=2
  u32x8 g1;
  g1[0] = 0x00020000u;                 // data_size=4B; mask/pad/iterate = 0
  g1[1] = 4096u << 16;                 // tensor_dim0 low16  (dim0 = 4096 elems)
  g1[2] = 0u | (1u << 16);             // tensor_dim0 hi16 | tensor_dim1 = 1
  g1[3] = 0u | (4096u << 16);          // tensor_dim1 hi16 | tile_dim0 = 4096
  g1[4] = 1u;                          // tile_dim1 = 1, tile_dim2 = 0
  g1[5] = 4096u;                       // tensor_dim0_stride low32
  g1[6] = 0u;                          // stride hi16 | dim1_stride low16
  g1[7] = 0u;                          // dim1_stride hi32
  asm volatile("tensor_load_to_lds %0, %1\n\t"
               "s_wait_tensorcnt 0x0"
               :: "s"(g0), "s"(g1)
               : "memory");
#else
  (void)gsrc; (void)staging;
#endif
}

// ---------------- token conv + positional embedding ----------------
// enc[b,l,n] = sum_{c,tap} x_enc[b,c,(l-1+tap) mod L] * w[n,c,tap] + PE[l,n]
__global__ void k_token_embed(const float* __restrict__ x,
                              const float* __restrict__ w,
                              float* __restrict__ enc) {
  int tid = blockIdx.x * blockDim.x + threadIdx.x;   // b*L*D + l*D + n
  int n  = tid & (kD - 1);
  int bl = tid >> 9;
  int l  = bl & (kL - 1);
  int b  = bl >> 9;
  int   i2  = n >> 1;
  float div = __expf(-(float)(2 * i2) * (9.210340371976184f / (float)kD));
  float ang = (float)l * div;
  float acc = (n & 1) ? __cosf(ang) : __sinf(ang);
  const float* xb = x + (size_t)b * kCIN * kL;
  #pragma unroll
  for (int tap = 0; tap < 3; ++tap) {
    int lp = (l + tap + kL - 1) & (kL - 1);
    const float* wr = w + (size_t)n * kCIN * 3 + tap;
    for (int c = 0; c < kCIN; ++c)
      acc += xb[c * kL + lp] * wr[c * 3];
  }
  enc[tid] = acc;
}

// ---------------- f32 -> bf16 convert ----------------
__global__ void k_cvt(const float* __restrict__ s, u16* __restrict__ d, int n) {
  int i = blockIdx.x * blockDim.x + threadIdx.x;
  if (i < n) d[i] = f2bf(s[i]);
}

// ---------------- WMMA bf16 GEMM: out = A(MxK) * W(NxK)^T + bias [epilogue] ----------------
// 128x128 block tile, 256 threads = 8 waves; wave (wr,wc) owns 32x64; BK = 32.
// Tile staging uses GLOBAL_LOAD_ASYNC_TO_LDS_B128 (ASYNCcnt) when available.
__global__ __launch_bounds__(256) void k_gemm(
    const u16* __restrict__ A, const u16* __restrict__ W,
    const float* __restrict__ bias, const float* __restrict__ resid,
    float* __restrict__ outF, u16* __restrict__ outH,
    int M, int N, int K, int act) {
  __shared__ __align__(16) u16 As[128][40];   // +8 pad keeps 16B alignment, breaks bank conflicts
  __shared__ __align__(16) u16 Ws[128][40];

  const int tid   = threadIdx.x;
  const int mBase = blockIdx.y * 128;
  const int nBase = blockIdx.x * 128;
  const int wave  = tid >> 5, lane = tid & 31;
  const int wr = wave >> 1, wc = wave & 1;
  const int half = lane >> 4, lr = lane & 15;

  v8f acc[2][4];
  const v8f vz = {0.f, 0.f, 0.f, 0.f, 0.f, 0.f, 0.f, 0.f};
  #pragma unroll
  for (int mi = 0; mi < 2; ++mi)
    #pragma unroll
    for (int ni = 0; ni < 4; ++ni) acc[mi][ni] = vz;

  // each thread copies 2 chunks of 8 bf16 for As and Ws (128 rows x 4 chunks)
  const int ch0 = tid * 2,     r0 = ch0 >> 2, c0 = (ch0 & 3) * 8;
  const int ch1 = tid * 2 + 1, r1 = ch1 >> 2, c1 = (ch1 & 3) * 8;

  union Fr { u32x4 q[2]; v16bf v; };

  for (int kk = 0; kk < K; kk += 32) {
#if HAVE_ASYNC_LDS
    async_cp16(&A[(size_t)(mBase + r0) * K + kk + c0], &As[r0][c0]);
    async_cp16(&A[(size_t)(mBase + r1) * K + kk + c1], &As[r1][c1]);
    async_cp16(&W[(size_t)(nBase + r0) * K + kk + c0], &Ws[r0][c0]);
    async_cp16(&W[(size_t)(nBase + r1) * K + kk + c1], &Ws[r1][c1]);
#else
    *(u32x4*)&As[r0][c0] = *(const u32x4*)&A[(size_t)(mBase + r0) * K + kk + c0];
    *(u32x4*)&As[r1][c1] = *(const u32x4*)&A[(size_t)(mBase + r1) * K + kk + c1];
    *(u32x4*)&Ws[r0][c0] = *(const u32x4*)&W[(size_t)(nBase + r0) * K + kk + c0];
    *(u32x4*)&Ws[r1][c1] = *(const u32x4*)&W[(size_t)(nBase + r1) * K + kk + c1];
#endif
    if (kk + 32 < K) {  // global_prefetch_b8 of next K-tile
      __builtin_prefetch(&A[(size_t)(mBase + r0) * K + kk + 32 + c0], 0, 0);
      __builtin_prefetch(&W[(size_t)(nBase + r0) * K + kk + 32 + c0], 0, 0);
    }
#if HAVE_ASYNC_LDS
    async_wait0();       // our async tile writes landed in LDS
#endif
    __syncthreads();

    Fr a[2], bf[4];
    #pragma unroll
    for (int mi = 0; mi < 2; ++mi) {
      int row = wr * 32 + mi * 16 + lr;
      a[mi].q[0] = *(const u32x4*)&As[row][half * 8];
      a[mi].q[1] = *(const u32x4*)&As[row][16 + half * 8];
    }
    #pragma unroll
    for (int ni = 0; ni < 4; ++ni) {
      int nr = wc * 64 + ni * 16 + lr;
      bf[ni].q[0] = *(const u32x4*)&Ws[nr][half * 8];
      bf[ni].q[1] = *(const u32x4*)&Ws[nr][16 + half * 8];
    }
    #pragma unroll
    for (int mi = 0; mi < 2; ++mi)
      #pragma unroll
      for (int ni = 0; ni < 4; ++ni)
        acc[mi][ni] = __builtin_amdgcn_wmma_f32_16x16x32_bf16(
            false, a[mi].v, false, bf[ni].v, (short)0, acc[mi][ni], false, false);
    __syncthreads();
  }

  // epilogue: bias, optional gelu, optional residual, f32 and/or bf16 stores
  #pragma unroll
  for (int mi = 0; mi < 2; ++mi) {
    #pragma unroll
    for (int ni = 0; ni < 4; ++ni) {
      int colg = nBase + wc * 64 + ni * 16 + lr;
      float bv = bias ? bias[colg] : 0.f;
      #pragma unroll
      for (int v = 0; v < 8; ++v) {
        int rowg = mBase + wr * 32 + mi * 16 + half * 8 + v;
        float val = acc[mi][ni][v] + bv;
        if (act == 1) val = gelu_f(val);
        if (resid) val += resid[(size_t)rowg * N + colg];
        if (outF) outF[(size_t)rowg * N + colg] = val;
        if (outH) outH[(size_t)rowg * N + colg] = f2bf(val);
      }
    }
  }
}

// ---------------- ProbSparse attention pieces ----------------
__global__ void k_sample_scores(const float* __restrict__ q, const float* __restrict__ kb,
                                float* __restrict__ Mout, int layer) {
  int tid = blockIdx.x * blockDim.x + threadIdx.x;   // bh*512 + l
  int l = tid & (kL - 1);
  int bh = tid >> 9;
  int h = bh & (kH - 1), b = bh >> 3;
  const float* qr = q + ((size_t)(b * kL + l)) * kD + h * kDh;
  float mx = -3.0e38f, sm = 0.f;
  for (int u = 0; u < kU; ++u) {
    int ki = hash3((u32)layer, (u32)l, (u32)u) & (kL - 1);
    const float* kr = kb + ((size_t)(b * kL + ki)) * kD + h * kDh;
    float d = 0.f;
    #pragma unroll 8
    for (int j = 0; j < kDh; ++j) d += qr[j] * kr[j];
    mx = fmaxf(mx, d); sm += d;
  }
  Mout[tid] = mx - sm * (1.f / (float)kL);
}

__global__ void k_topk(const float* __restrict__ Min, int* __restrict__ top) {
  __shared__ float sv[512];
  __shared__ float rv[256];
  __shared__ int   ri[256];
  int bh = blockIdx.x, t = threadIdx.x;
  sv[t] = Min[bh * kL + t];
  sv[t + 256] = Min[bh * kL + t + 256];
  __syncthreads();
  for (int it = 0; it < kU; ++it) {
    float v0 = sv[t], v1 = sv[t + 256];
    float bvv = v0; int bii = t;
    if (v1 > bvv) { bvv = v1; bii = t + 256; }
    rv[t] = bvv; ri[t] = bii;
    __syncthreads();
    for (int s = 128; s > 0; s >>= 1) {
      if (t < s && rv[t + s] > rv[t]) { rv[t] = rv[t + s]; ri[t] = ri[t + s]; }
      __syncthreads();
    }
    if (t == 0) { top[bh * kU + it] = ri[0]; sv[ri[0]] = -3.0e38f; }
    __syncthreads();
  }
}

__global__ void k_qk_scores(const float* __restrict__ q, const float* __restrict__ kb,
                            const int* __restrict__ top, float* __restrict__ sc) {
  int tid = blockIdx.x * blockDim.x + threadIdx.x;   // (bh*35+u)*512 + kc
  int kc = tid & (kL - 1);
  int r = tid >> 9;
  int u = r % kU, bh = r / kU;
  int h = bh & (kH - 1), b = bh >> 3;
  int lq = top[bh * kU + u];
  const float* qr = q + ((size_t)(b * kL + lq)) * kD + h * kDh;
  const float* kr = kb + ((size_t)(b * kL + kc)) * kD + h * kDh;
  float d = 0.f;
  #pragma unroll 8
  for (int j = 0; j < kDh; ++j) d += qr[j] * kr[j];
  sc[tid] = d * 0.125f;
}

__global__ void k_softmax(float* __restrict__ sc) {
  __shared__ float red[256];
  int row = blockIdx.x, t = threadIdx.x;
  float* r = sc + (size_t)row * kL;
  float v0 = r[t], v1 = r[t + 256];
  red[t] = fmaxf(v0, v1);
  __syncthreads();
  for (int s = 128; s > 0; s >>= 1) { if (t < s) red[t] = fmaxf(red[t], red[t + s]); __syncthreads(); }
  float m = red[0];
  __syncthreads();
  float e0 = __expf(v0 - m), e1 = __expf(v1 - m);
  red[t] = e0 + e1;
  __syncthreads();
  for (int s = 128; s > 0; s >>= 1) { if (t < s) red[t] += red[t + s]; __syncthreads(); }
  float inv = 1.f / red[0];
  r[t] = e0 * inv; r[t + 256] = e1 * inv;
}

__global__ void k_vmean(const float* __restrict__ v, float* __restrict__ vm) {
  int tid = blockIdx.x * blockDim.x + threadIdx.x;   // bh*64 + d
  int d = tid & (kDh - 1);
  int bh = tid >> 6;
  int h = bh & (kH - 1), b = bh >> 3;
  float s = 0.f;
  for (int l = 0; l < kL; ++l) s += v[((size_t)(b * kL + l)) * kD + h * kDh + d];
  vm[tid] = s * (1.f / (float)kL);
}

__global__ void k_ctx_fill(const float* __restrict__ vm, float* __restrict__ ctx) {
  int tid = blockIdx.x * blockDim.x + threadIdx.x;   // (b*L+l)*512 + hd
  int hd = tid & (kD - 1);
  int bl = tid >> 9;
  int b = bl >> 9;
  int h = hd >> 6, d = hd & (kDh - 1);
  ctx[tid] = vm[((size_t)(b * kH + h)) * kDh + d];
}

__global__ void k_upd_scatter(const float* __restrict__ sc, const float* __restrict__ v,
                              const int* __restrict__ top, float* __restrict__ ctx) {
  int tid = blockIdx.x * blockDim.x + threadIdx.x;   // (bh*35+u)*64 + d
  int d = tid & (kDh - 1);
  int r = tid >> 6;
  int u = r % kU, bh = r / kU;
  int h = bh & (kH - 1), b = bh >> 3;
  const float* srow = sc + (size_t)(bh * kU + u) * kL;
  float acc = 0.f;
  for (int kc = 0; kc < kL; ++kc)
    acc += srow[kc] * v[((size_t)(b * kL + kc)) * kD + h * kDh + d];
  int l = top[bh * kU + u];
  ctx[((size_t)(b * kL + l)) * kD + h * kDh + d] = acc;
}

// ---------------- layernorm over D=512 per row ----------------
__global__ void k_layernorm(const float* __restrict__ x, const float* __restrict__ g,
                            const float* __restrict__ bb, float* __restrict__ o) {
  __shared__ float s1[256], s2[256];
  int row = blockIdx.x, t = threadIdx.x;
  const float* xr = x + (size_t)row * kD;
  float x0 = xr[t], x1 = xr[t + 256];
  s1[t] = x0 + x1; s2[t] = x0 * x0 + x1 * x1;
  __syncthreads();
  for (int s = 128; s > 0; s >>= 1) { if (t < s) { s1[t] += s1[t + s]; s2[t] += s2[t + s]; } __syncthreads(); }
  float mu = s1[0] * (1.f / (float)kD);
  float var = s2[0] * (1.f / (float)kD) - mu * mu;
  float rs = rsqrtf(var + 1e-5f);
  float* orow = o + (size_t)row * kD;
  orow[t]       = (x0 - mu) * rs * g[t]       + bb[t];
  orow[t + 256] = (x1 - mu) * rs * g[t + 256] + bb[t + 256];
}

// ---------------- classifier: logits[b,c] = sum_k gelu(x[b,k]) * W[c,k] + b[c] ----------------
__global__ void k_logits(const float* __restrict__ x, const float* __restrict__ w,
                         const float* __restrict__ bb, float* __restrict__ out) {
  __shared__ float red[256];
  int c = blockIdx.x, b = blockIdx.y, t = threadIdx.x;
  const int KK = kL * kD;
  const float* xr = x + (size_t)b * KK;
  const float* wr = w + (size_t)c * KK;
  float acc = 0.f;
  for (int kk = t; kk < KK; kk += 256) acc += gelu_f(xr[kk]) * wr[kk];
  red[t] = acc;
  __syncthreads();
  for (int s = 128; s > 0; s >>= 1) { if (t < s) red[t] += red[t + s]; __syncthreads(); }
  if (t == 0) out[b * kCLS + c] = red[0] + bb[c];
}

// ---------------- loss = -mean_b log_softmax(logits)[label] ----------------
__global__ void k_loss(const float* __restrict__ logits, const int* __restrict__ label,
                       float* __restrict__ loss) {
  __shared__ float s[kB];
  int t = threadIdx.x;
  if (t < kB) {
    const float* lg = logits + t * kCLS;
    float m = -3.0e38f;
    for (int c = 0; c < kCLS; ++c) m = fmaxf(m, lg[c]);
    float se = 0.f;
    for (int c = 0; c < kCLS; ++c) se += __expf(lg[c] - m);
    int lab = label[t];
    s[t] = -(lg[lab] - m - __logf(se));
  }
  __syncthreads();
  if (t == 0) {
    float tot = 0.f;
    for (int i = 0; i < kB; ++i) tot += s[i];
    loss[0] = tot * (1.f / (float)kB);
  }
}

// ---------------- host orchestration ----------------
extern "C" void kernel_launch(void* const* d_in, const int* in_sizes, int n_in,
                              void* d_out, int out_size, void* d_ws, size_t ws_size,
                              hipStream_t stream) {
  (void)in_sizes; (void)n_in; (void)out_size; (void)ws_size;
  const float* x_enc  = (const float*)d_in[0];
  const int*   label  = (const int*)d_in[1];
  const float* tokw   = (const float*)d_in[2];
  auto LPf = [&](int li, int j) -> const float* { return (const float*)d_in[3 + li * 16 + j]; };
  const float* norm_g = (const float*)d_in[51];
  const float* norm_b = (const float*)d_in[52];
  const float* cls_w  = (const float*)d_in[53];
  const float* cls_b  = (const float*)d_in[54];

  char* ws = (char*)d_ws;
  float* enc  = (float*)(ws + 0);                       // 33.5 MB
  float* tmp  = (float*)(ws + 33554432ull);             // ctx / normed
  float* qb   = (float*)(ws + 67108864ull);
  float* kbf  = (float*)(ws + 100663296ull);
  float* vb   = (float*)(ws + 134217728ull);
  float* ao   = (float*)(ws + 167772160ull);            // gemm f32 out
  u16*   x16  = (u16*)(ws + 201326592ull);              // 16384x512 bf16
  u16*   f16  = (u16*)(ws + 218103808ull);              // 16384x2048 bf16
  u16*   w16  = (u16*)(ws + 285212672ull);              // weight bf16 (max 2048x512)
  float* Mbuf = (float*)(ws + 287309824ull);            // 256x512
  int*   top  = (int*)(ws + 287834112ull);              // 256x35
  float* sc   = (float*)(ws + 287870976ull);            // 256x35x512
  float* vm   = (float*)(ws + 306221056ull);            // 256x64
  float* out  = (float*)d_out;                          // 320 logits + 1 loss

  auto cvt = [&](const float* s, u16* d, int n) {
    k_cvt<<<(n + 255) / 256, 256, 0, stream>>>(s, d, n);
  };
  auto gemm = [&](const u16* Ax, const u16* Wx, const float* bias, const float* resid,
                  float* oF, u16* oH, int Mm, int Nn, int Kk, int act) {
    dim3 g(Nn / 128, Mm / 128);
    k_gemm<<<g, 256, 0, stream>>>(Ax, Wx, bias, resid, oF, oH, Mm, Nn, Kk, act);
  };

  // token conv + PE
  k_token_embed<<<(kM * kD) / 256, 256, 0, stream>>>(x_enc, tokw, enc);

  for (int li = 0; li < 3; ++li) {
    const float *Wq = LPf(li, 0),  *bq = LPf(li, 1);
    const float *Wk = LPf(li, 2),  *bk = LPf(li, 3);
    const float *Wv = LPf(li, 4),  *bv = LPf(li, 5);
    const float *Wo = LPf(li, 6),  *bo = LPf(li, 7);
    const float *c1w = LPf(li, 8), *c1b = LPf(li, 9);
    const float *c2w = LPf(li, 10), *c2b = LPf(li, 11);
    const float *g1 = LPf(li, 12), *b1 = LPf(li, 13);
    const float *g2 = LPf(li, 14), *b2 = LPf(li, 15);

    // QKV projections (bf16 WMMA, f32 accumulate)
    cvt(enc, x16, kM * kD);
    cvt(Wq, w16, kD * kD);  gemm(x16, w16, bq, nullptr, qb,  nullptr, kM, kD, kD, 0);
    cvt(Wk, w16, kD * kD);  gemm(x16, w16, bk, nullptr, kbf, nullptr, kM, kD, kD, 0);
    cvt(Wv, w16, kD * kD);  gemm(x16, w16, bv, nullptr, vb,  nullptr, kM, kD, kD, 0);

    // ProbSparse attention
    k_sample_scores<<<(kB * kH * kL) / 256, 256, 0, stream>>>(qb, kbf, Mbuf, li);
    k_topk<<<kB * kH, 256, 0, stream>>>(Mbuf, top);
    k_qk_scores<<<(kB * kH * kU * kL) / 256, 256, 0, stream>>>(qb, kbf, top, sc);
    k_softmax<<<kB * kH * kU, 256, 0, stream>>>(sc);
    k_vmean<<<(kB * kH * kDh) / 256, 256, 0, stream>>>(vb, vm);
    k_ctx_fill<<<(kM * kD) / 256, 256, 0, stream>>>(vm, tmp);
    k_upd_scatter<<<(kB * kH * kU * kDh) / 256, 256, 0, stream>>>(sc, vb, top, tmp);

    // O projection + residual, then LN1
    cvt(tmp, x16, kM * kD);
    cvt(Wo, w16, kD * kD);  gemm(x16, w16, bo, enc, ao, nullptr, kM, kD, kD, 0);
    k_layernorm<<<kM, 256, 0, stream>>>(ao, g1, b1, enc);

    // FFN: TDM-warm W1 through L2, then gelu(x W1^T + b1) W2^T + b2 + resid, LN2
    k_tdm_preload<<<(kFF * kD * 4) / 16384, 32, 0, stream>>>((const u32*)c1w);
    cvt(enc, x16, kM * kD);
    cvt(c1w, w16, kFF * kD); gemm(x16, w16, c1b, nullptr, nullptr, f16, kM, kFF, kD, 1);
    cvt(c2w, w16, kD * kFF); gemm(f16, w16, c2b, enc, ao, nullptr, kM, kD, kFF, 0);
    k_layernorm<<<kM, 256, 0, stream>>>(ao, g2, b2, enc);
  }

  // final norm -> gelu (fused in classifier) -> logits -> loss
  k_layernorm<<<kM, 256, 0, stream>>>(enc, norm_g, norm_b, tmp);
  k_tdm_preload<<<(kCLS * kL * kD * 4) / 16384, 32, 0, stream>>>((const u32*)cls_w);
  dim3 lg(kCLS, kB);
  k_logits<<<lg, 256, 0, stream>>>(tmp, cls_w, cls_b, out);
  k_loss<<<1, 32, 0, stream>>>(out, label, out + kB * kCLS);
}
